// GlobalAttention_27324581937528
// MI455X (gfx1250) — compile-verified, matrix operand order checked by reading, not run on hardware
//
#include <hip/hip_runtime.h>
#include <hip/hip_bf16.h>

// Problem constants (from reference): B=16, Lq=Lc=2048, D=1024
#define B_    16
#define LQ    2048
#define LC    2048
#define D_    1024
#define NROWS (B_ * LQ)                      // 32768 flattened (b,q) rows
#define OUT_ELEMS ((size_t)NROWS * D_)       // output tensor elements (f32)
#define WPAD  80                             // LDS weight-tile row stride (bf16 elems)

typedef __attribute__((ext_vector_type(16))) __bf16 v16bf;
typedef __attribute__((ext_vector_type(8)))  float  v8f;

union BF16Frag {
  v16bf v;
  unsigned short u[16];
  unsigned uu[8];
  uint4 q[2];
};

// Two f32 -> packed bf16 pair with a single v_perm_b32 (truncation).
__device__ __forceinline__ unsigned pack_bf16x2(float lo, float hi) {
  return __builtin_amdgcn_perm(__float_as_uint(hi), __float_as_uint(lo), 0x07060302u);
}

// Scalar f32 -> bf16 (round-to-nearest-even) for epilogue stores only.
__device__ __forceinline__ unsigned short f2bf(float f) {
  union { float f; unsigned u; } x; x.f = f;
  unsigned r = x.u + 0x7FFFu + ((x.u >> 16) & 1u);
  return (unsigned short)(r >> 16);
}

// D = A*B + C, bf16 inputs, f32 accum
__device__ __forceinline__ v8f wmma_bf16(v16bf a, v16bf b, v8f c) {
  return __builtin_amdgcn_wmma_f32_16x16x32_bf16(false, a, false, b,
                                                 (short)0, c, false, false);
}

// Row-major bf16 source (global or LDS): two b128 loads per lane.
// ISA 16-bit operand striping: lanes 0-15 hold K={0..7,16..23}, lanes 16-31 K={8..15,24..31}.
__device__ __forceinline__ v16bf frag_from_bf16(const unsigned short* p, int hi) {
  BF16Frag a;
  a.q[0] = *reinterpret_cast<const uint4*>(p + hi * 8);
  a.q[1] = *reinterpret_cast<const uint4*>(p + 16 + hi * 8);
  return a.v;
}

// Row-major f32 source, converted with 8 v_perm ops (4x b128 loads).
__device__ __forceinline__ v16bf frag_from_f32(const float* p, int hi) {
  BF16Frag a;
  float4 x0 = *reinterpret_cast<const float4*>(p + hi * 8);
  float4 x1 = *reinterpret_cast<const float4*>(p + hi * 8 + 4);
  float4 y0 = *reinterpret_cast<const float4*>(p + 16 + hi * 8);
  float4 y1 = *reinterpret_cast<const float4*>(p + 16 + hi * 8 + 4);
  a.uu[0] = pack_bf16x2(x0.x, x0.y);
  a.uu[1] = pack_bf16x2(x0.z, x0.w);
  a.uu[2] = pack_bf16x2(x1.x, x1.y);
  a.uu[3] = pack_bf16x2(x1.z, x1.w);
  a.uu[4] = pack_bf16x2(y0.x, y0.y);
  a.uu[5] = pack_bf16x2(y0.z, y0.w);
  a.uu[6] = pack_bf16x2(y1.x, y1.y);
  a.uu[7] = pack_bf16x2(y1.z, y1.w);
  return a.v;
}

// gfx1250 async global->LDS copy (16B per lane), tracked by ASYNCcnt.
__device__ __forceinline__ void async_g2l_b128(unsigned lds_off, const unsigned short* g) {
  asm volatile("global_load_async_to_lds_b128 %0, %1, off"
               :: "v"(lds_off), "v"(g) : "memory");
}
__device__ __forceinline__ void wait_async0() {
  asm volatile("s_wait_asynccnt 0x0" ::: "memory");
}

// ---------------- Kernel 0a: elementwise f32 -> bf16 (weight tables) ----------------
__global__ __launch_bounds__(256)
void k_cvt(const float* __restrict__ src, unsigned* __restrict__ dst, int n8) {
  int t = blockIdx.x * 256 + threadIdx.x;
  if (t >= n8) return;
  const float4* s = reinterpret_cast<const float4*>(src) + 2 * (size_t)t;
  float4 a = s[0], b = s[1];
  uint4 o;
  o.x = pack_bf16x2(a.x, a.y);
  o.y = pack_bf16x2(a.z, a.w);
  o.z = pack_bf16x2(b.x, b.y);
  o.w = pack_bf16x2(b.z, b.w);
  reinterpret_cast<uint4*>(dst)[t] = o;
}

// ---------------- Kernel 0b: bank[b][k][d] f32 -> bankT[b][d][k] bf16 (LDS tiled) ----------------
__global__ __launch_bounds__(256)
void k_transpose_bf16(const float* __restrict__ bank, unsigned short* __restrict__ bankT) {
  __shared__ alignas(4) unsigned short tile[64][66];
  const int tid = threadIdx.x;
  const int k0 = blockIdx.x * 64;
  const int d0 = blockIdx.y * 64;
  const int b  = blockIdx.z;
  const float* src = bank + ((size_t)b * LC + k0) * D_ + d0;
#pragma unroll
  for (int i = 0; i < 4; ++i) {           // 64x64 f32 in, float4 per thread
    int e  = i * 256 + tid;
    int kk = e >> 4;
    int dd = (e & 15) * 4;
    float4 v = *reinterpret_cast<const float4*>(src + (size_t)kk * D_ + dd);
    unsigned* w = reinterpret_cast<unsigned*>(&tile[kk][dd]);
    w[0] = pack_bf16x2(v.x, v.y);
    w[1] = pack_bf16x2(v.z, v.w);
  }
  __syncthreads();
#pragma unroll
  for (int i = 0; i < 8; ++i) {           // write transposed, 2 bf16 per store
    int e  = i * 256 + tid;
    int dd = e >> 5;
    int kk = (e & 31) * 2;
    unsigned v = (unsigned)tile[kk][dd] | ((unsigned)tile[kk + 1][dd] << 16);
    *reinterpret_cast<unsigned*>(&bankT[((size_t)b * D_ + d0 + dd) * LC + k0 + kk]) = v;
  }
}

// ---------------- Kernel 1: ctx_proj[b,k,e] = sum_d bank[b,k,d] * Wc[e,d] -> bf16 ----------------
// Weight chunk (64 cols x 64 d, bf16) staged to LDS via async DMA, double-buffered,
// shared by all 8 waves. B fragments preloaded as an array so ds_loads clause together.
__global__ __launch_bounds__(256)
void k_ctx_proj(const float* __restrict__ bank, const unsigned short* __restrict__ Wch,
                unsigned short* __restrict__ ctxh) {
  __shared__ alignas(16) unsigned short wt[2][64][WPAD];
  const int tid  = threadIdx.x;
  const int wave = tid >> 5, lane = tid & 31;
  const int l = lane & 15, hi = lane >> 4;
  const int m0 = blockIdx.y * 128 + wave * 16;   // flattened (b,k) row tile
  const int n0 = blockIdx.x * 64;                // e tile
  const float* arow = bank + (size_t)(m0 + l) * D_;

  auto stage = [&](int buf, int kc) {
#pragma unroll
    for (int s = 0; s < 2; ++s) {                // 64 rows x 8 segs = 512 x 16B, 2 per thread
      int u = s * 256 + tid;
      int row = u >> 3, seg = u & 7;
      unsigned loff = (unsigned)(size_t)(&wt[buf][row][seg * 8]);
      async_g2l_b128(loff, Wch + (size_t)(n0 + row) * D_ + kc + seg * 8);
    }
  };

  v8f acc[4] = {};
  stage(0, 0);
  wait_async0();
  __syncthreads();
  int buf = 0;
  for (int kc = 0; kc < D_; kc += 64) {
    if (kc + 64 < D_) stage(buf ^ 1, kc + 64);   // overlap DMA with WMMAs
#pragma unroll
    for (int half = 0; half < 2; ++half) {
      const int k0 = kc + half * 32;
      v16bf a = frag_from_f32(arow + k0, hi);
      v16bf bfr[4];
#pragma unroll
      for (int nt = 0; nt < 4; ++nt)
        bfr[nt] = frag_from_bf16(&wt[buf][nt * 16 + l][half * 32], hi);  // ds_load clause
#pragma unroll
      for (int nt = 0; nt < 4; ++nt)
        acc[nt] = wmma_bf16(a, bfr[nt], acc[nt]);
    }
    wait_async0();
    __syncthreads();
    buf ^= 1;
  }
#pragma unroll
  for (int nt = 0; nt < 4; ++nt)
#pragma unroll
    for (int r = 0; r < 8; ++r)
      ctxh[(size_t)(m0 + hi * 8 + r) * D_ + (n0 + nt * 16 + l)] = f2bf(acc[nt][r]);
}

// ------- Kernel 2: raw scores (Q @ ctx^T / sqrt(D)) + online row max / sum-of-exp stats -------
// 32 query rows per block (64 KB bf16 Q in LDS); each wave computes a 2x2 tile grid per pass
// so every ctx B-fragment pair feeds 4 WMMAs.
__global__ __launch_bounds__(256)
void k_score_stats(const float* __restrict__ query, const unsigned short* __restrict__ ctxh,
                   float* __restrict__ attn, float* __restrict__ statsM,
                   float* __restrict__ statsL) {
  __shared__ alignas(16) unsigned short qs[32 * D_];  // 64 KB bf16 Q tile (reused for stats reduce)
  const int tid  = threadIdx.x;
  const int wave = tid >> 5, lane = tid & 31;
  const int l = lane & 15, hi = lane >> 4;
  const int q0 = blockIdx.x * 32;            // flattened (b,q) row base
  const int b  = q0 / LQ;

  // Stage Q tile to LDS as bf16 (float4 loads, v_perm pack, b128 LDS stores)
#pragma unroll 4
  for (int i = 0; i < (32 * D_) / (256 * 8); ++i) {
    int e = (i * 256 + tid) * 8;
    int row = e >> 10, col = e & (D_ - 1);
    const float4* s = reinterpret_cast<const float4*>(query + (size_t)(q0 + row) * D_ + col);
    float4 a = s[0], bq = s[1];
    uint4 o;
    o.x = pack_bf16x2(a.x, a.y);
    o.y = pack_bf16x2(a.z, a.w);
    o.z = pack_bf16x2(bq.x, bq.y);
    o.w = pack_bf16x2(bq.z, bq.w);
    *reinterpret_cast<uint4*>(&qs[e]) = o;
  }
  __syncthreads();

  float Mrun[2][8], Lrun[2][8];
#pragma unroll
  for (int qt = 0; qt < 2; ++qt)
#pragma unroll
    for (int r = 0; r < 8; ++r) { Mrun[qt][r] = -3.0e38f; Lrun[qt][r] = 0.0f; }

  const unsigned short* qrow0 = qs + l * D_;
  const unsigned short* qrow1 = qs + (16 + l) * D_;
  for (int pass = 0; pass < 8; ++pass) {     // 8 waves x 8 passes x 2 ktiles = 128 k-tiles
    const int kbase = (pass * 8 + wave) * 32;
    const unsigned short* crow0 = ctxh + (size_t)(b * LC + kbase + l) * D_;
    const unsigned short* crow1 = crow0 + (size_t)16 * D_;
    v8f acc[2][2] = {};
    for (int d0 = 0; d0 < D_; d0 += 32) {
      v16bf a0 = frag_from_bf16(qrow0 + d0, hi);     // ds_load_b128 x2
      v16bf a1 = frag_from_bf16(qrow1 + d0, hi);
      v16bf c0 = frag_from_bf16(crow0 + d0, hi);     // global b128 x2
      v16bf c1 = frag_from_bf16(crow1 + d0, hi);
      acc[0][0] = wmma_bf16(a0, c0, acc[0][0]);
      acc[0][1] = wmma_bf16(a0, c1, acc[0][1]);
      acc[1][0] = wmma_bf16(a1, c0, acc[1][0]);
      acc[1][1] = wmma_bf16(a1, c1, acc[1][1]);
    }
#pragma unroll
    for (int qt = 0; qt < 2; ++qt)
#pragma unroll
      for (int kt = 0; kt < 2; ++kt) {
        const int kcol = kbase + kt * 16 + l;
#pragma unroll
        for (int r = 0; r < 8; ++r) {
          float s = acc[qt][kt][r] * 0.03125f;   // 1/sqrt(1024)
          attn[(size_t)(q0 + qt * 16 + hi * 8 + r) * LC + kcol] = s;   // raw score
          float v = s;
          v = fmaxf(v, __shfl_xor(v, 1, 32));
          v = fmaxf(v, __shfl_xor(v, 2, 32));
          v = fmaxf(v, __shfl_xor(v, 4, 32));
          v = fmaxf(v, __shfl_xor(v, 8, 32));
          float newM = fmaxf(Mrun[qt][r], v);
          float e = __expf(s - newM);
          e += __shfl_xor(e, 1, 32);
          e += __shfl_xor(e, 2, 32);
          e += __shfl_xor(e, 4, 32);
          e += __shfl_xor(e, 8, 32);
          Lrun[qt][r] = Lrun[qt][r] * __expf(Mrun[qt][r] - newM) + e;
          Mrun[qt][r] = newM;
        }
      }
  }

  // Cross-wave reduce: overlay stats arrays onto the Q LDS (all reads of qs are done).
  __syncthreads();
  float* redM = reinterpret_cast<float*>(qs);          // [8][32]
  float* redL = redM + 8 * 32;
  if (l == 0) {
#pragma unroll
    for (int qt = 0; qt < 2; ++qt)
#pragma unroll
      for (int r = 0; r < 8; ++r) {
        redM[wave * 32 + qt * 16 + hi * 8 + r] = Mrun[qt][r];
        redL[wave * 32 + qt * 16 + hi * 8 + r] = Lrun[qt][r];
      }
  }
  __syncthreads();
  if (tid < 32) {
    float m = -3.0e38f;
#pragma unroll
    for (int w = 0; w < 8; ++w) m = fmaxf(m, redM[w * 32 + tid]);
    float s = 0.0f;
#pragma unroll
    for (int w = 0; w < 8; ++w) s += redL[w * 32 + tid] * __expf(redM[w * 32 + tid] - m);
    statsM[q0 + tid] = m;
    statsL[q0 + tid] = s;
  }
}

// ---- Kernel 3a: softmax finalize attn in place (pure streaming, one block per row) ----
__global__ __launch_bounds__(256)
void k_attn_norm(float* __restrict__ attn, const float* __restrict__ statsM,
                 const float* __restrict__ statsL) {
  const int row = blockIdx.x;
  const float m = statsM[row];
  const float rinv = 1.0f / statsL[row];
  float* p = attn + (size_t)row * LC;
#pragma unroll
  for (int i = 0; i < 2; ++i) {
    int c = (i * 256 + threadIdx.x) * 4;
    float4 s = *reinterpret_cast<const float4*>(p + c);
    float4 o;
    o.x = __expf(s.x - m) * rinv;
    o.y = __expf(s.y - m) * rinv;
    o.z = __expf(s.z - m) * rinv;
    o.w = __expf(s.w - m) * rinv;
    *reinterpret_cast<float4*>(p + c) = o;
  }
}

// ---- Kernel 3b: context = attn @ bank -> bf16 (no LDS; 32 q rows x 512 e cols per block) ----
// A fragments from normalized f32 attn (v_perm pack, shared by 4 e-tiles);
// B fragments preloaded as an array (global b128 clause), shared by 2 q-tiles.
__global__ __launch_bounds__(256)
void k_context(const unsigned short* __restrict__ bankT, const float* __restrict__ attn,
               unsigned short* __restrict__ cth) {
  const int tid  = threadIdx.x;
  const int wave = tid >> 5, lane = tid & 31;
  const int l = lane & 15, hi = lane >> 4;
  const int q0  = blockIdx.y * 32;
  const int e0w = blockIdx.x * 512 + wave * 64;   // 4 e-tiles of 16 per wave
  const int b   = q0 / LQ;
  const float* arow0 = attn + (size_t)(q0 + l) * LC;
  const float* arow1 = attn + (size_t)(q0 + 16 + l) * LC;
  v8f acc[2][4] = {};
  for (int k0 = 0; k0 < LC; k0 += 32) {
    v16bf a0 = frag_from_f32(arow0 + k0, hi);
    v16bf a1 = frag_from_f32(arow1 + k0, hi);
    v16bf bb[4];
#pragma unroll
    for (int j = 0; j < 4; ++j)
      bb[j] = frag_from_bf16(bankT + ((size_t)b * D_ + e0w + j * 16 + l) * LC + k0, hi);
#pragma unroll
    for (int j = 0; j < 4; ++j) {
      acc[0][j] = wmma_bf16(a0, bb[j], acc[0][j]);
      acc[1][j] = wmma_bf16(a1, bb[j], acc[1][j]);
    }
  }
#pragma unroll
  for (int qt = 0; qt < 2; ++qt)
#pragma unroll
    for (int j = 0; j < 4; ++j)
#pragma unroll
      for (int r = 0; r < 8; ++r)
        cth[(size_t)(q0 + qt * 16 + hi * 8 + r) * D_ + e0w + j * 16 + l] = f2bf(acc[qt][j][r]);
}

// ---- Kernel 4: out = context @ Woc^T + query @ Woq^T (both weight chunks async-staged) ----
__global__ __launch_bounds__(256)
void k_output(const float* __restrict__ query, const unsigned short* __restrict__ cth,
              const unsigned short* __restrict__ Woqh, const unsigned short* __restrict__ Woch,
              float* __restrict__ out) {
  __shared__ alignas(16) unsigned short wtC[2][64][WPAD];
  __shared__ alignas(16) unsigned short wtQ[2][64][WPAD];
  const int tid  = threadIdx.x;
  const int wave = tid >> 5, lane = tid & 31;
  const int l = lane & 15, hi = lane >> 4;
  const int m0 = blockIdx.y * 128 + wave * 16;
  const int n0 = blockIdx.x * 64;
  const float* qrow          = query + (size_t)(m0 + l) * D_;
  const unsigned short* crow = cth   + (size_t)(m0 + l) * D_;

  auto stage = [&](int buf, int kc) {
#pragma unroll
    for (int s = 0; s < 2; ++s) {
      int u = s * 256 + tid;
      int row = u >> 3, seg = u & 7;
      unsigned offC = (unsigned)(size_t)(&wtC[buf][row][seg * 8]);
      unsigned offQ = (unsigned)(size_t)(&wtQ[buf][row][seg * 8]);
      const size_t g = (size_t)(n0 + row) * D_ + kc + seg * 8;
      async_g2l_b128(offC, Woch + g);
      async_g2l_b128(offQ, Woqh + g);
    }
  };

  v8f acc[4] = {};
  stage(0, 0);
  wait_async0();
  __syncthreads();
  int buf = 0;
  for (int kc = 0; kc < D_; kc += 64) {
    if (kc + 64 < D_) stage(buf ^ 1, kc + 64);
#pragma unroll
    for (int half = 0; half < 2; ++half) {
      const int k0 = kc + half * 32;
      v16bf aC = frag_from_bf16(crow + k0, hi);
      v16bf aQ = frag_from_f32(qrow + k0, hi);
      v16bf bC[4], bQ[4];
#pragma unroll
      for (int nt = 0; nt < 4; ++nt) {
        bC[nt] = frag_from_bf16(&wtC[buf][nt * 16 + l][half * 32], hi);
        bQ[nt] = frag_from_bf16(&wtQ[buf][nt * 16 + l][half * 32], hi);
      }
#pragma unroll
      for (int nt = 0; nt < 4; ++nt) {
        acc[nt] = wmma_bf16(aC, bC[nt], acc[nt]);
        acc[nt] = wmma_bf16(aQ, bQ[nt], acc[nt]);
      }
    }
    wait_async0();
    __syncthreads();
    buf ^= 1;
  }
#pragma unroll
  for (int nt = 0; nt < 4; ++nt)
#pragma unroll
    for (int r = 0; r < 8; ++r)
      out[(size_t)(m0 + hi * 8 + r) * D_ + n0 + nt * 16 + l] = acc[nt][r];
}

extern "C" void kernel_launch(void* const* d_in, const int* in_sizes, int n_in,
                              void* d_out, int out_size, void* d_ws, size_t ws_size,
                              hipStream_t stream) {
  (void)in_sizes; (void)n_in; (void)out_size; (void)ws_size;
  const float* query = (const float*)d_in[0];
  const float* bank  = (const float*)d_in[1];
  const float* Wc    = (const float*)d_in[2];
  const float* Woq   = (const float*)d_in[3];
  const float* Woc   = (const float*)d_in[4];

  float* out  = (float*)d_out;
  float* attn = out + OUT_ELEMS;               // second tuple element, flat after output

  // Workspace layout (bytes):
  //   bankT 64MB | ctxh 64MB | cth 64MB | Wch 2MB | Woqh 2MB | Woch 2MB | stats 256KB
  char* ws = (char*)d_ws;
  const size_t SZ_BIG = (size_t)NROWS * D_ * 2;          // 67,108,864 B
  const size_t SZ_W   = (size_t)D_ * D_ * 2;             // 2,097,152 B
  unsigned short* bankT  = (unsigned short*)(ws);
  unsigned short* ctxh   = (unsigned short*)(ws + SZ_BIG);
  unsigned short* cth    = (unsigned short*)(ws + 2 * SZ_BIG);
  unsigned short* Wch    = (unsigned short*)(ws + 3 * SZ_BIG);
  unsigned short* Woqh   = (unsigned short*)(ws + 3 * SZ_BIG + SZ_W);
  unsigned short* Woch   = (unsigned short*)(ws + 3 * SZ_BIG + 2 * SZ_W);
  float*          statsM = (float*)(ws + 3 * SZ_BIG + 3 * SZ_W);
  float*          statsL = statsM + NROWS;

  dim3 blk(256);
  const int wN8 = (D_ * D_) / 8;
  k_cvt<<<dim3(wN8 / 256), blk, 0, stream>>>(Wc,  (unsigned*)Wch,  wN8);
  k_cvt<<<dim3(wN8 / 256), blk, 0, stream>>>(Woq, (unsigned*)Woqh, wN8);
  k_cvt<<<dim3(wN8 / 256), blk, 0, stream>>>(Woc, (unsigned*)Woch, wN8);
  k_transpose_bf16<<<dim3(LC / 64, D_ / 64, B_), blk, 0, stream>>>(bank, bankT);

  k_ctx_proj    <<<dim3(D_ / 64, NROWS / 128), blk, 0, stream>>>(bank, Wch, ctxh);
  k_score_stats <<<dim3(NROWS / 32),           blk, 0, stream>>>(query, ctxh, attn, statsM, statsL);
  k_attn_norm   <<<dim3(NROWS),                blk, 0, stream>>>(attn, statsM, statsL);
  k_context     <<<dim3(D_ / 512, NROWS / 32), blk, 0, stream>>>(bankT, attn, cth);
  k_output      <<<dim3(D_ / 64, NROWS / 128), blk, 0, stream>>>(query, cth, Woqh, Woch, out);
}
